// GCN5_MN_TANH_67980742361106
// MI455X (gfx1250) — compile-verified, matrix operand order checked by reading, not run on hardware
//
#include <hip/hip_runtime.h>
#include <math.h>

// ---------------------------------------------------------------------------
// GCN (5x GraphConv + mean-pool + MLP) for MI455X / gfx1250.
//  - SpMM scatter: global f32 atomics (memory-bound path, ~13GB/layer)
//  - Dense [N,256]x[256,256] layers: v_wmma_f32_16x16x32_bf16, f32 accum,
//    k-loop fully unrolled so loads pipeline ahead of WMMAs
//  - norm_dst scaling + f32->bf16 (v_cvt_pk_bf16_f32) fused into A-frag build
//  - intermediate layer outputs stored pre-scaled by norm_src (saves a pass)
// ---------------------------------------------------------------------------

#define N_NODES  200000
#define N_EDGES  6400000
#define N_GRAPHS 1024
#define HDIM     256

typedef __attribute__((ext_vector_type(16))) __bf16 v16bf;
typedef __attribute__((ext_vector_type(2)))  __bf16 v2bf;
typedef __attribute__((ext_vector_type(8)))  float  v8f;

__device__ __forceinline__ unsigned short f2bf(float f) {
  unsigned int u = __float_as_uint(f);
  u += 0x7FFFu + ((u >> 16) & 1u);          // round-to-nearest-even
  return (unsigned short)(u >> 16);
}

// Pack two f32 -> packed bf16x2 (one v_cvt_pk_bf16_f32 when available).
__device__ __forceinline__ unsigned int pack2bf(float lo, float hi) {
#if __has_builtin(__builtin_amdgcn_cvt_pk_bf16_f32)
  v2bf r = __builtin_amdgcn_cvt_pk_bf16_f32(lo, hi);
  union { v2bf v; unsigned int u; } c;
  c.v = r;
  return c.u;
#else
  return (unsigned int)f2bf(lo) | ((unsigned int)f2bf(hi) << 16);
#endif
}

// ------------------------------- utility -----------------------------------
__global__ void fill_zero(float* __restrict__ p, int n) {
  int i = blockIdx.x * blockDim.x + threadIdx.x;
  if (i < n) p[i] = 0.0f;
}

// ------------------------------ degrees ------------------------------------
__global__ void degree_kernel(const int* __restrict__ src,
                              const int* __restrict__ dst,
                              float* __restrict__ deg_in,
                              float* __restrict__ deg_out) {
  int e = blockIdx.x * blockDim.x + threadIdx.x;
  if (e < N_EDGES) {
    unsafeAtomicAdd(&deg_out[src[e]], 1.0f);
    unsafeAtomicAdd(&deg_in [dst[e]], 1.0f);
  }
}

// ---------------- node features (pre-scaled by norm_src) -------------------
__global__ void node_kernel(const float* __restrict__ deg_in,
                            const float* __restrict__ deg_out,
                            float* __restrict__ norm_src,
                            float* __restrict__ norm_dst,
                            float* __restrict__ h4s) {
  int n = blockIdx.x * blockDim.x + threadIdx.x;
  if (n < N_NODES) {
    float di = deg_in[n], dq = deg_out[n];
    float ns = rsqrtf(fmaxf(dq, 1.0f));
    float nd = rsqrtf(fmaxf(di, 1.0f));
    norm_src[n] = ns;
    norm_dst[n] = nd;
    float h1 = di;
    float h2 = (di - 3.0f > 0.0f) ? 1.0f : 0.0f;
    float h3 = 3.0f / di;
    float h4 = (di - 4.0f > 0.0f) ? 1.0f : 0.0f;
    h4s[n * 4 + 0] = h1 * ns;
    h4s[n * 4 + 1] = h2 * ns;
    h4s[n * 4 + 2] = h3 * ns;
    h4s[n * 4 + 3] = h4 * ns;
  }
}

// ------------------------------- SpMM --------------------------------------
__global__ void spmm4(const int* __restrict__ src, const int* __restrict__ dst,
                      const float* __restrict__ xs4, float* __restrict__ agg4) {
  int e = blockIdx.x * blockDim.x + threadIdx.x;
  if (e < N_EDGES) {
    int s = src[e], d = dst[e];
    float4 v = *(const float4*)&xs4[(size_t)s * 4];
    float* p = &agg4[(size_t)d * 4];
    unsafeAtomicAdd(p + 0, v.x);
    unsafeAtomicAdd(p + 1, v.y);
    unsafeAtomicAdd(p + 2, v.z);
    unsafeAtomicAdd(p + 3, v.w);
  }
}

__global__ void spmm256(const int* __restrict__ src, const int* __restrict__ dst,
                        const float* __restrict__ xs, float* __restrict__ agg) {
  long t = (long)blockIdx.x * blockDim.x + threadIdx.x;   // E * 64 threads
  if (t < (long)N_EDGES * 64) {
    int e = (int)(t >> 6);
    int c = ((int)t & 63) << 2;                 // 4 floats per thread
    int s = src[e], d = dst[e];
    float4 v = *(const float4*)&xs[(long)s * HDIM + c];
    float* p = &agg[(long)d * HDIM + c];
    unsafeAtomicAdd(p + 0, v.x);
    unsafeAtomicAdd(p + 1, v.y);
    unsafeAtomicAdd(p + 2, v.z);
    unsafeAtomicAdd(p + 3, v.w);
  }
}

// ---------------- layer 1: [N,4] @ [4,256], tanh, pre-scale ----------------
__global__ void dense1(const float* __restrict__ agg4,
                       const float* __restrict__ norm_dst,
                       const float* __restrict__ norm_src,
                       const float* __restrict__ W1, const float* __restrict__ b1,
                       float* __restrict__ out) {
  int t = blockIdx.x * blockDim.x + threadIdx.x;
  if (t < N_NODES * HDIM) {
    int n = t >> 8, c = t & 255;
    float nd = norm_dst[n];
    const float* a = &agg4[(size_t)n * 4];
    float acc = b1[c];
#pragma unroll
    for (int k = 0; k < 4; k++) acc += (a[k] * nd) * W1[k * HDIM + c];
    out[t] = tanhf(acc) * norm_src[n];
  }
}

// ------------- W [K,Nc] f32 -> W^T [Nc,K] bf16 (for WMMA B frags) ----------
__global__ void wcvt(const float* __restrict__ W, unsigned short* __restrict__ WT) {
  int t = blockIdx.x * blockDim.x + threadIdx.x;
  if (t < HDIM * HDIM) {
    int nn = t & 255, k = t >> 8;
    WT[nn * HDIM + k] = f2bf(W[k * HDIM + nn]);       // coalesced read of W
  }
}

// --------- WMMA GEMM: tanh((A*nd) @ W + b) [* norm_src]  -------------------
// One wave computes a 16-row x 64-col tile (4 accumulators, 32 WMMAs).
// 4 waves/block cover all 256 columns of one 16-row tile.
__global__ __launch_bounds__(128) void gemm_tanh_wmma(
    const float* __restrict__ A,               // [N,256] f32 (agg)
    const unsigned short* __restrict__ WT,     // [256,256] bf16, WT[n][k]=W[k][n]
    const float* __restrict__ bias,            // [256]
    const float* __restrict__ norm_dst,        // [N]
    const float* __restrict__ post_scale,      // norm_src or nullptr
    float* __restrict__ out) {                 // [N,256]
  const int lane = threadIdx.x & 31;
  const int wave = threadIdx.x >> 5;
  const int rowTile = blockIdx.x;
  const int lr  = lane & 15;
  const int sel = lane >> 4;                   // 0: lanes 0-15, 1: lanes 16-31
  const int row = rowTile * 16 + lr;
  const float nd = norm_dst[row];
  const int colbase0 = wave * 64;

  v8f acc[4] = {};
  const float* arow = A + (long)row * HDIM;

#pragma unroll
  for (int k0 = 0; k0 < HDIM; k0 += 32) {
    // --- A fragment: lane holds K = k0+8*sel .. +7  and  k0+16+8*sel .. +7
    union { v16bf v; unsigned int w[8]; } af;
    const float4* p0 = (const float4*)(arow + k0 + 8 * sel);
    const float4* p1 = (const float4*)(arow + k0 + 16 + 8 * sel);
    float4 f0 = p0[0], f1 = p0[1], f2 = p1[0], f3 = p1[1];
    af.w[0] = pack2bf(f0.x * nd, f0.y * nd);
    af.w[1] = pack2bf(f0.z * nd, f0.w * nd);
    af.w[2] = pack2bf(f1.x * nd, f1.y * nd);
    af.w[3] = pack2bf(f1.z * nd, f1.w * nd);
    af.w[4] = pack2bf(f2.x * nd, f2.y * nd);
    af.w[5] = pack2bf(f2.z * nd, f2.w * nd);
    af.w[6] = pack2bf(f3.x * nd, f3.y * nd);
    af.w[7] = pack2bf(f3.z * nd, f3.w * nd);

#pragma unroll
    for (int j = 0; j < 4; j++) {
      const int n = colbase0 + j * 16 + lr;    // B column held by this lane
      union { v16bf v; uint4 q[2]; } bfr;
      const unsigned short* wrow = WT + (long)n * HDIM;
      bfr.q[0] = *(const uint4*)(wrow + k0 + 8 * sel);
      bfr.q[1] = *(const uint4*)(wrow + k0 + 16 + 8 * sel);
      acc[j] = __builtin_amdgcn_wmma_f32_16x16x32_bf16(
          false, af.v, false, bfr.v, (short)0, acc[j], false, false);
    }
  }

  // C/D layout: VGPR g -> row rowTile*16 + g + 8*sel, col = colbase + lr
#pragma unroll
  for (int j = 0; j < 4; j++) {
    const int col = colbase0 + j * 16 + lr;
    const float bb = bias[col];
#pragma unroll
    for (int g = 0; g < 8; g++) {
      const int r = rowTile * 16 + g + 8 * sel;
      float v = tanhf(acc[j][g] + bb);
      if (post_scale) v *= post_scale[r];
      out[(long)r * HDIM + col] = v;
    }
  }
}

// ------------------------------ pooling ------------------------------------
__global__ void pool_sum(const float* __restrict__ h, const int* __restrict__ gid,
                         float* __restrict__ gsum) {
  long t = (long)blockIdx.x * blockDim.x + threadIdx.x;   // N * 64
  if (t < (long)N_NODES * 64) {
    int n = (int)(t >> 6);
    int c = ((int)t & 63) << 2;
    int g = gid[n];
    float4 v = *(const float4*)&h[(long)n * HDIM + c];
    float* p = &gsum[(long)g * HDIM + c];
    unsafeAtomicAdd(p + 0, v.x);
    unsafeAtomicAdd(p + 1, v.y);
    unsafeAtomicAdd(p + 2, v.z);
    unsafeAtomicAdd(p + 3, v.w);
  }
}

__global__ void pool_cnt(const int* __restrict__ gid, float* __restrict__ cnt) {
  int n = blockIdx.x * blockDim.x + threadIdx.x;
  if (n < N_NODES) unsafeAtomicAdd(&cnt[gid[n]], 1.0f);
}

__global__ void pool_final(const float* __restrict__ gsum,
                           const float* __restrict__ cnt,
                           float* __restrict__ emb) {
  int t = blockIdx.x * blockDim.x + threadIdx.x;
  if (t < N_GRAPHS * HDIM) {
    int g = t >> 8;
    float c = cnt[g];
    emb[t] = (c > 0.0f) ? gsum[t] / fmaxf(c, 1.0f) : 0.0f;
  }
}

// ------------------------------ MLP head -----------------------------------
__global__ void mlp1(const float* __restrict__ emb, const float* __restrict__ Wl1,
                     const float* __restrict__ bl1, float* __restrict__ e1) {
  int t = blockIdx.x * blockDim.x + threadIdx.x;
  if (t < N_GRAPHS * (HDIM / 2)) {
    int g = t >> 7, j = t & 127;
    const float* er = &emb[(size_t)g * HDIM];
    float acc = bl1[j];
    for (int k = 0; k < HDIM; k++) acc += er[k] * Wl1[k * (HDIM / 2) + j];
    e1[t] = (acc > 0.0f) ? acc : 0.01f * acc;   // leaky_relu(0.01)
  }
}

__global__ void mlp2(const float* __restrict__ e1, const float* __restrict__ Wl2,
                     const float* __restrict__ bl2, float* __restrict__ pred) {
  int g = blockIdx.x * blockDim.x + threadIdx.x;
  if (g < N_GRAPHS) {
    const float* er = &e1[(size_t)g * (HDIM / 2)];
    float acc = bl2[0];
    for (int k = 0; k < HDIM / 2; k++) acc += er[k] * Wl2[k];
    pred[g] = 1.0f / (1.0f + expf(-acc));
  }
}

// ---------------------------------------------------------------------------
extern "C" void kernel_launch(void* const* d_in, const int* in_sizes, int n_in,
                              void* d_out, int out_size, void* d_ws, size_t ws_size,
                              hipStream_t stream) {
  (void)in_sizes; (void)n_in; (void)out_size; (void)ws_size;
  const int*   src = (const int*)d_in[0];
  const int*   dst = (const int*)d_in[1];
  const int*   gid = (const int*)d_in[2];
  const float* W1  = (const float*)d_in[3];  const float* b1  = (const float*)d_in[4];
  const float* W2  = (const float*)d_in[5];  const float* b2  = (const float*)d_in[6];
  const float* W3  = (const float*)d_in[7];  const float* b3  = (const float*)d_in[8];
  const float* W4  = (const float*)d_in[9];  const float* b4  = (const float*)d_in[10];
  const float* W5  = (const float*)d_in[11]; const float* b5  = (const float*)d_in[12];
  const float* Wl1 = (const float*)d_in[13]; const float* bl1 = (const float*)d_in[14];
  const float* Wl2 = (const float*)d_in[15]; const float* bl2 = (const float*)d_in[16];

  // outputs: pred [G], graph_emb [G,H], h [N,H] (flat, in return order)
  float* pred = (float*)d_out;
  float* emb  = pred + N_GRAPHS;
  float* hout = emb + (size_t)N_GRAPHS * HDIM;

  // ---- carve workspace (256B aligned) ----
  char* w = (char*)d_ws;
  auto carve = [&](size_t bytes) -> void* {
    void* p = (void*)w;
    w += (bytes + 255) & ~(size_t)255;
    return p;
  };
  float* deg_in   = (float*)carve((size_t)2 * N_NODES * 4);   // deg_in + deg_out
  float* deg_out  = deg_in + N_NODES;
  float* norm_src = (float*)carve((size_t)N_NODES * 4);
  float* norm_dst = (float*)carve((size_t)N_NODES * 4);
  float* h4s      = (float*)carve((size_t)N_NODES * 4 * 4);
  float* agg4     = (float*)carve((size_t)N_NODES * 4 * 4);
  float* bufA     = (float*)carve((size_t)N_NODES * HDIM * 4);
  float* bufB     = (float*)carve((size_t)N_NODES * HDIM * 4);
  unsigned short* WT[4];
  for (int i = 0; i < 4; i++) WT[i] = (unsigned short*)carve((size_t)HDIM * HDIM * 2);
  float* gsum = (float*)carve((size_t)N_GRAPHS * HDIM * 4 + N_GRAPHS * 4);  // gsum + cnt
  float* cnt  = gsum + (size_t)N_GRAPHS * HDIM;
  float* e1   = (float*)carve((size_t)N_GRAPHS * (HDIM / 2) * 4);

  const int T = 256;
  auto nb = [](long n, int t) { return (int)((n + t - 1) / t); };

  // degrees + node features
  fill_zero<<<nb(2L * N_NODES, T), T, 0, stream>>>(deg_in, 2 * N_NODES);
  degree_kernel<<<nb(N_EDGES, T), T, 0, stream>>>(src, dst, deg_in, deg_out);
  node_kernel<<<nb(N_NODES, T), T, 0, stream>>>(deg_in, deg_out, norm_src, norm_dst, h4s);

  // weight transpose+bf16 (once)
  wcvt<<<nb(HDIM * HDIM, T), T, 0, stream>>>(W2, WT[0]);
  wcvt<<<nb(HDIM * HDIM, T), T, 0, stream>>>(W3, WT[1]);
  wcvt<<<nb(HDIM * HDIM, T), T, 0, stream>>>(W4, WT[2]);
  wcvt<<<nb(HDIM * HDIM, T), T, 0, stream>>>(W5, WT[3]);

  // layer 1: SpMM on 4-wide features, then dense [4->256]
  fill_zero<<<nb(4L * N_NODES, T), T, 0, stream>>>(agg4, 4 * N_NODES);
  spmm4<<<nb(N_EDGES, T), T, 0, stream>>>(src, dst, h4s, agg4);
  dense1<<<nb((long)N_NODES * HDIM, T), T, 0, stream>>>(agg4, norm_dst, norm_src, W1, b1, bufA);

  // layers 2..5: SpMM (f32 atomics) + WMMA GEMM (+tanh, +pre-scale)
  const float* biases[4] = {b2, b3, b4, b5};
  for (int L = 0; L < 4; L++) {
    fill_zero<<<nb((long)N_NODES * HDIM, T), T, 0, stream>>>(bufB, N_NODES * HDIM);
    spmm256<<<nb((long)N_EDGES * 64, T), T, 0, stream>>>(src, dst, bufA, bufB);
    bool last = (L == 3);
    gemm_tanh_wmma<<<N_NODES / 16, 128, 0, stream>>>(
        bufB, WT[L], biases[L], norm_dst,
        last ? (const float*)nullptr : norm_src,
        last ? hout : bufA);
  }

  // mean-pool per graph
  fill_zero<<<nb((long)N_GRAPHS * HDIM + N_GRAPHS, T), T, 0, stream>>>(
      gsum, N_GRAPHS * HDIM + N_GRAPHS);
  pool_sum<<<nb((long)N_NODES * 64, T), T, 0, stream>>>(hout, gid, gsum);
  pool_cnt<<<nb(N_NODES, T), T, 0, stream>>>(gid, cnt);
  pool_final<<<nb(N_GRAPHS * HDIM, T), T, 0, stream>>>(gsum, cnt, emb);

  // MLP head
  mlp1<<<nb(N_GRAPHS * (HDIM / 2), T), T, 0, stream>>>(emb, Wl1, bl1, e1);
  mlp2<<<nb(N_GRAPHS, T), T, 0, stream>>>(e1, Wl2, bl2, pred);
}
